// MultiheadAttention_32255204393561
// MI455X (gfx1250) — matrix operand order repaired and checked by reading, not compiled, not run on hardware
//
#include <hip/hip_runtime.h>

// ---------------------------------------------------------------------------
// MI455X (gfx1250, wave32) fused MHA: bf16 WMMA GEMMs + flash attention.
// Softmax scale (1/sqrt(D) * log2(e)) is folded into the Q projection so the
// attention inner loop runs in log2-domain; row reductions use DPP16 row_ror
// (pure VALU) instead of ds_bpermute round-trips.
// ---------------------------------------------------------------------------

#define DEV __device__ __forceinline__

typedef __attribute__((ext_vector_type(16))) __bf16       bf16x16;
typedef __attribute__((ext_vector_type(2)))  __bf16       bf16x2;
typedef __attribute__((ext_vector_type(8)))  float        f32x8;
typedef __attribute__((ext_vector_type(4)))  unsigned int u32x4;

constexpr int Bc = 2, Sc = 2048, Ec = 1024, Hc = 16, Dc = 64;

union FragU {
  bf16x16        bf;
  u32x4          q[2];
  unsigned short us[16];
};

DEV unsigned short bfbits(float x) {   // fp32 -> bf16 (hardware convert)
  __bf16 h = (__bf16)x;
  unsigned short u;
  __builtin_memcpy(&u, &h, 2);
  return u;
}

DEV unsigned int pack_bf16x2(float lo, float hi) {
#if __has_builtin(__builtin_amdgcn_cvt_pk_bf16_f32)
  bf16x2 r = __builtin_amdgcn_cvt_pk_bf16_f32(lo, hi);
  unsigned int u;
  __builtin_memcpy(&u, &r, 4);
  return u;
#else
  return (unsigned)bfbits(lo) | ((unsigned)bfbits(hi) << 16);
#endif
}

// ---- DPP16 row reductions: 16-lane halves of a wave32 (all lanes get result)
template <int CTRL>
DEV float dpp_f32(float x) {
  return __int_as_float(__builtin_amdgcn_update_dpp(
      0, __float_as_int(x), CTRL, /*row_mask=*/0xF, /*bank_mask=*/0xF,
      /*bound_ctrl=*/true));
}
DEV float red_max16(float x) {  // row_ror:1,2,4,8
  x = fmaxf(x, dpp_f32<0x121>(x));
  x = fmaxf(x, dpp_f32<0x122>(x));
  x = fmaxf(x, dpp_f32<0x124>(x));
  x = fmaxf(x, dpp_f32<0x128>(x));
  return x;
}
DEV float red_sum16(float x) {
  x += dpp_f32<0x121>(x);
  x += dpp_f32<0x122>(x);
  x += dpp_f32<0x124>(x);
  x += dpp_f32<0x128>(x);
  return x;
}

DEV f32x8 wmma_bf16(bf16x16 a, bf16x16 b, f32x8 c) {
  // D = A(16x32 bf16) * B(32x16 bf16) + C(16x16 f32)
  return __builtin_amdgcn_wmma_f32_16x16x32_bf16(
      /*neg_a=*/false, a, /*neg_b=*/false, b,
      /*c_mod=*/(short)0, c, /*reuse_a=*/false, /*reuse_b=*/false);
}

// A fragment 16x32 (rows m, contraction K contiguous in memory).
// ISA 7.12.2: lanes 0-15 rows 0-15; VGPR0-3 <-> K = hi*8+0..7,
// VGPR4-7 <-> K = 16+hi*8+0..7  => two 16-byte loads per lane.
DEV bf16x16 load_frag_a(const unsigned short* base, int ld, int m0, int k0) {
  const int lane = threadIdx.x & 31;
  const int m = lane & 15, hi = lane >> 4;
  const unsigned short* p = base + (long)(m0 + m) * ld + k0 + hi * 8;
  FragU f;
  f.q[0] = *(const u32x4*)(p);
  f.q[1] = *(const u32x4*)(p + 16);
  return f.bf;
}

// B fragment 32x16: source stored [N, K] row-major (column n of B == row n of
// source), contraction contiguous. Lane: n = lane&15, VGPR0-3 <-> K = hi*8+0..7,
// VGPR4-7 <-> K = 16+hi*8+0..7  => two 16-byte loads per lane.
DEV bf16x16 load_frag_b(const unsigned short* base, int ld, int n0, int k0) {
  const int lane = threadIdx.x & 31;
  const int n = lane & 15, hi = lane >> 4;
  const unsigned short* p = base + (long)(n0 + n) * ld + k0 + hi * 8;
  FragU f;
  f.q[0] = *(const u32x4*)(p);
  f.q[1] = *(const u32x4*)(p + 16);
  return f.bf;
}

// ---------------------------------------------------------------------------
// fp32 -> bf16 conversion pass (2 elements / iteration, packed dword stores)
// ---------------------------------------------------------------------------
__global__ void cvt_f32_to_bf16(const float* __restrict__ in,
                                unsigned int* __restrict__ out, int n2) {
  int i = blockIdx.x * blockDim.x + threadIdx.x;
  int stride = gridDim.x * blockDim.x;
  for (; i < n2; i += stride) {
    float2 f = ((const float2*)in)[i];
    out[i] = pack_bf16x2(f.x, f.y);
  }
}

// ---------------------------------------------------------------------------
// C[M,N] = scale * (A[M,K] * B[N,K]^T)   (bf16 in, f32 accumulate)
// One wave computes a 64x64 tile: 16 WMMAs per K-step of 32.
// MODE 0: store bf16 row-major [M,N]
// MODE 1: store f32  row-major [M,N]
// MODE 2: store bf16 transposed as Vt[(b*H+h)*D + d][S]  (n = h*64+d, m = b*S+s)
// ---------------------------------------------------------------------------
template <int MODE>
__global__ __launch_bounds__(32) void gemm_bf16_nt(
    const unsigned short* __restrict__ A, const unsigned short* __restrict__ Bm,
    void* __restrict__ Cout, int M, int N, int K, float scale) {
  const int n0 = blockIdx.x * 64;
  const int m0 = blockIdx.y * 64;
  const int lane = threadIdx.x & 31;
  const int nn = lane & 15, hi = lane >> 4;

  f32x8 acc[4][4] = {};
  for (int k0 = 0; k0 < K; k0 += 32) {
    if (k0 + 32 < K) {  // hint next K-slab toward the cache hierarchy
      __builtin_prefetch(A + (long)m0 * K + k0 + 32, 0, 1);
      __builtin_prefetch(Bm + (long)n0 * K + k0 + 32, 0, 1);
    }
    bf16x16 a[4], b[4];
#pragma unroll
    for (int t = 0; t < 4; ++t) a[t] = load_frag_a(A, K, m0 + t * 16, k0);
#pragma unroll
    for (int t = 0; t < 4; ++t) b[t] = load_frag_b(Bm, K, n0 + t * 16, k0);
#pragma unroll
    for (int mt = 0; mt < 4; ++mt)
#pragma unroll
      for (int nt = 0; nt < 4; ++nt)
        acc[mt][nt] = wmma_bf16(a[mt], b[nt], acc[mt][nt]);
  }

  if (MODE == 2) {
    // transpose on store: per lane, rows v=0..7 are 8 consecutive s values ->
    // pack into one 16-byte store along S.
#pragma unroll
    for (int mt = 0; mt < 4; ++mt)
#pragma unroll
      for (int nt = 0; nt < 4; ++nt) {
        int n = n0 + nt * 16 + nn;       // n = h*64 + d
        int h = n >> 6, d = n & 63;
        int mbase = m0 + mt * 16 + 8 * hi;
        int bb = mbase >> 11;            // batch   (m / 2048)
        int s = mbase & 2047;            // seq pos
        u32x4 pk;
        pk.x = pack_bf16x2(acc[mt][nt][0] * scale, acc[mt][nt][1] * scale);
        pk.y = pack_bf16x2(acc[mt][nt][2] * scale, acc[mt][nt][3] * scale);
        pk.z = pack_bf16x2(acc[mt][nt][4] * scale, acc[mt][nt][5] * scale);
        pk.w = pack_bf16x2(acc[mt][nt][6] * scale, acc[mt][nt][7] * scale);
        *(u32x4*)((unsigned short*)Cout +
                  ((long)(bb * Hc + h) * Dc + d) * Sc + s) = pk;
      }
  } else {
#pragma unroll
    for (int mt = 0; mt < 4; ++mt)
#pragma unroll
      for (int nt = 0; nt < 4; ++nt)
#pragma unroll
        for (int v = 0; v < 8; ++v) {
          long m = m0 + mt * 16 + v + 8 * hi;
          long n = n0 + nt * 16 + nn;
          if (MODE == 0)
            ((unsigned short*)Cout)[m * N + n] = bfbits(acc[mt][nt][v] * scale);
          else
            ((float*)Cout)[m * N + n] = acc[mt][nt][v] * scale;
        }
  }
}

// ---------------------------------------------------------------------------
// Flash attention: one wave per (b, h, 16-query tile). Online softmax.
// Qp is pre-scaled by (1/sqrt(D))*log2(e), so scores are log2-domain logits:
//   p = exp2(s - m),  correction = exp2(m_old - m_new).
// Qp,Kp: bf16 [B*S, E] row-major; Vt: bf16 [B*H, D, S]; Ao: bf16 [B*S, E].
// ---------------------------------------------------------------------------
__global__ __launch_bounds__(32) void flash_attn16(
    const unsigned short* __restrict__ Qp, const unsigned short* __restrict__ Kp,
    const unsigned short* __restrict__ Vt, unsigned short* __restrict__ Ao) {
  __shared__ __align__(16) unsigned short Plds[16 * 64];

  const int q0 = blockIdx.x * 16;
  const int h = blockIdx.y;
  const int b = blockIdx.z;
  const int lane = threadIdx.x & 31;
  const int nn = lane & 15, hi = lane >> 4;

  const unsigned short* Qbase = Qp + (long)b * Sc * Ec + (long)h * Dc;
  const unsigned short* Kbase = Kp + (long)b * Sc * Ec + (long)h * Dc;
  const unsigned short* Vbase = Vt + (long)(b * Hc + h) * Dc * Sc;

  // Q tile 16x64 as two A-fragments (d = 0..31, 32..63); held in regs all loop
  bf16x16 aq0 = load_frag_a(Qbase, Ec, q0, 0);
  bf16x16 aq1 = load_frag_a(Qbase, Ec, q0, 32);

  f32x8 O[4] = {};          // 16 x 64 f32 accumulator (4 C/D tiles)
  float mr[8], lr[8];       // per-lane row stats (C/D layout: row = v + 8*hi)
#pragma unroll
  for (int v = 0; v < 8; ++v) { mr[v] = -1e30f; lr[v] = 0.f; }

  for (int kb = 0; kb < Sc; kb += 64) {
    // ---- log2-domain logits S = (cQ) K^T for 16 x 64 key block (8 WMMAs) --
    f32x8 sc[4];
#pragma unroll
    for (int nt = 0; nt < 4; ++nt) {
      f32x8 z = {};
      bf16x16 bk0 = load_frag_b(Kbase, Ec, kb + nt * 16, 0);
      bf16x16 bk1 = load_frag_b(Kbase, Ec, kb + nt * 16, 32);
      sc[nt] = wmma_bf16(aq1, bk1, wmma_bf16(aq0, bk0, z));
    }

    // ---- per-row block max (DPP reduction across each 16-lane half) ----
    float bm[8];
#pragma unroll
    for (int v = 0; v < 8; ++v)
      bm[v] = red_max16(
          fmaxf(fmaxf(sc[0][v], sc[1][v]), fmaxf(sc[2][v], sc[3][v])));

    // ---- online-softmax rescale of running state ----
#pragma unroll
    for (int v = 0; v < 8; ++v) {
      float mn = fmaxf(mr[v], bm[v]);
      float corr = __builtin_exp2f(mr[v] - mn);
      mr[v] = mn;
      lr[v] *= corr;
#pragma unroll
      for (int dt = 0; dt < 4; ++dt) O[dt][v] *= corr;
    }

    // ---- probabilities + row sums ----
    float bs[8];
#pragma unroll
    for (int v = 0; v < 8; ++v) bs[v] = 0.f;
#pragma unroll
    for (int nt = 0; nt < 4; ++nt)
#pragma unroll
      for (int v = 0; v < 8; ++v) {
        float p = __builtin_exp2f(sc[nt][v] - mr[v]);
        sc[nt][v] = p;
        bs[v] += p;
      }
#pragma unroll
    for (int v = 0; v < 8; ++v) lr[v] += red_sum16(bs[v]);

    // ---- D-layout -> A-layout via LDS round trip (packed conversions) ----
    __syncthreads();  // WAR vs previous iteration's ds_loads
#pragma unroll
    for (int nt = 0; nt < 4; ++nt)
#pragma unroll
      for (int v = 0; v < 8; v += 2) {
        unsigned int pk = pack_bf16x2(sc[nt][v], sc[nt][v + 1]);
        Plds[(v + 8 * hi) * 64 + nt * 16 + nn] = (unsigned short)(pk & 0xffffu);
        Plds[(v + 1 + 8 * hi) * 64 + nt * 16 + nn] = (unsigned short)(pk >> 16);
      }
    __syncthreads();
    asm volatile("s_wait_dscnt 0" ::: "memory");

    // ---- O += P V  (8 WMMAs; V is pre-transposed so K is contiguous) ----
#pragma unroll
    for (int kk = 0; kk < 2; ++kk) {
      bf16x16 pa = load_frag_a(Plds, 64, 0, kk * 32);
#pragma unroll
      for (int dt = 0; dt < 4; ++dt) {
        bf16x16 vb = load_frag_b(Vbase, Sc, dt * 16, kb + kk * 32);
        O[dt] = wmma_bf16(pa, vb, O[dt]);
      }
    }
  }

  // ---- normalize and store attention output (bf16, row-major [B*S, E]) ----
#pragma unroll
  for (int v = 0; v < 8; ++v) {
    float inv = 1.0f / lr[v];
#pragma unroll
    for (int dt = 0; dt < 4; ++dt) {
      long m = (long)b * Sc + q0 + v + 8 * hi;
      long n = (long)h * Dc + dt * 16 + nn;
      Ao[m * Ec + n] = bfbits(O[dt][v] * inv);
    }
  }
}

// ---------------------------------------------------------------------------
extern "C" void kernel_launch(void* const* d_in, const int* in_sizes, int n_in,
                              void* d_out, int out_size, void* d_ws,
                              size_t ws_size, hipStream_t stream) {
  (void)in_sizes; (void)n_in; (void)out_size; (void)ws_size;
  const float* q  = (const float*)d_in[0];
  const float* k  = (const float*)d_in[1];
  const float* v  = (const float*)d_in[2];
  const float* Wq = (const float*)d_in[3];
  const float* Wk = (const float*)d_in[4];
  const float* Wv = (const float*)d_in[5];
  const float* Wo = (const float*)d_in[6];
  float* out = (float*)d_out;

  const size_t NX = (size_t)Bc * Sc * Ec;  // 4 Mi activation elements
  const size_t NW = (size_t)Ec * Ec;       // 1 Mi weight elements
  unsigned short* ws = (unsigned short*)d_ws;  // total 32 Mi elems = 64 MB
  unsigned short* Xq = ws;
  unsigned short* Xk = Xq + NX;
  unsigned short* Xv = Xk + NX;
  unsigned short* Bq = Xv + NX;
  unsigned short* Bk = Bq + NW;
  unsigned short* Bv = Bk + NW;
  unsigned short* Bo = Bv + NW;
  unsigned short* Qp = Bo + NW;
  unsigned short* Kp = Qp + NX;
  unsigned short* Vt = Kp + NX;
  unsigned short* Ao = Vt + NX;

  cvt_f32_to_bf16<<<2048, 256, 0, stream>>>(q,  (unsigned int*)Xq, (int)(NX / 2));
  cvt_f32_to_bf16<<<2048, 256, 0, stream>>>(k,  (unsigned int*)Xk, (int)(NX / 2));
  cvt_f32_to_bf16<<<2048, 256, 0, stream>>>(v,  (unsigned int*)Xv, (int)(NX / 2));
  cvt_f32_to_bf16<<<1024, 256, 0, stream>>>(Wq, (unsigned int*)Bq, (int)(NW / 2));
  cvt_f32_to_bf16<<<1024, 256, 0, stream>>>(Wk, (unsigned int*)Bk, (int)(NW / 2));
  cvt_f32_to_bf16<<<1024, 256, 0, stream>>>(Wv, (unsigned int*)Bv, (int)(NW / 2));
  cvt_f32_to_bf16<<<1024, 256, 0, stream>>>(Wo, (unsigned int*)Bo, (int)(NW / 2));

  const float QSCALE = 0.125f * 1.44269504088896f;  // 1/sqrt(D) * log2(e)
  dim3 gg(Ec / 64, (Bc * Sc) / 64);  // 16 x 64 waves per GEMM
  gemm_bf16_nt<0><<<gg, 32, 0, stream>>>(Xq, Bq, Qp, Bc * Sc, Ec, Ec, QSCALE);
  gemm_bf16_nt<0><<<gg, 32, 0, stream>>>(Xk, Bk, Kp, Bc * Sc, Ec, Ec, 1.0f);
  gemm_bf16_nt<2><<<gg, 32, 0, stream>>>(Xv, Bv, Vt, Bc * Sc, Ec, Ec, 1.0f);

  dim3 ga(Sc / 16, Hc, Bc);          // 4096 waves
  flash_attn16<<<ga, 32, 0, stream>>>(Qp, Kp, Vt, Ao);

  gemm_bf16_nt<1><<<gg, 32, 0, stream>>>(Ao, Bo, out, Bc * Sc, Ec, Ec, 1.0f);
}
